// DAWN_88124138979400
// MI455X (gfx1250) — compile-verified, matrix operand order checked by reading, not run on hardware
//
#include <hip/hip_runtime.h>
#include <math.h>

// Problem constants (from reference)
constexpr int TOK  = 2048;   // B*S
constexpr int Dm   = 768;    // D
constexpr int Rr   = 128;    // R
constexpr int Nn   = 128;    // NF == NR
constexpr int DSd  = 64;     // DS
constexpr int TOPK = 4;      // K

typedef __attribute__((ext_vector_type(16))) __bf16 v16bf;
typedef __attribute__((ext_vector_type(8)))  float  v8f;

__device__ __forceinline__ v8f wmma_bf16(v16bf a, v16bf b, v8f c) {
  // 8 args: (neg_a, A, neg_b, B, c_mod, C, reuse_a, reuse_b)
  return __builtin_amdgcn_wmma_f32_16x16x32_bf16(false, a, false, b, (short)0, c, false, false);
}

// fragment swizzle for 16-bit A/B 16x32 tiles (CDNA5 ISA 7.12.2):
//   lane = (m or n) + 16*((K%16)/8),  elem e = (K/16)*8 + (K%8)
__device__ __forceinline__ int frag_lane(int mn, int K) { return mn + (((K >> 3) & 1) << 4); }
__device__ __forceinline__ int frag_elem(int K)         { return ((K >> 4) << 3) | (K & 7); }

// ---------------------------------------------------------------- utilities
__global__ void zero_u32(unsigned int* p, int n) {
  int i = blockIdx.x * blockDim.x + threadIdx.x;
  for (; i < n; i += gridDim.x * blockDim.x) p[i] = 0u;
}

// dst[t*ncols + c] = sum_{j<4} part[(t*4+j)*ncols + c]   (fixed order -> deterministic)
__global__ void reduce4_kernel(const float* __restrict__ part, float* __restrict__ dst, int ncols) {
  long long i = (long long)blockIdx.x * blockDim.x + threadIdx.x;
  long long total = (long long)TOK * ncols;
  if (i >= total) return;
  long long t = i / ncols, c = i % ncols;
  const float* p = part + (t * 4) * (long long)ncols + c;
  dst[i] = ((p[0] + p[(long long)ncols]) + p[2LL * ncols]) + p[3LL * ncols];
}

// ---------------------------------------------------------------- router
__global__ __launch_bounds__(128)
void router_kernel(const float* __restrict__ x,
                   const float* __restrict__ W_fk, const float* __restrict__ b_fk,
                   const float* __restrict__ W_rk, const float* __restrict__ b_rk,
                   const float* __restrict__ emb_fk, const float* __restrict__ emb_rk,
                   int* __restrict__ idx_f, float* __restrict__ w_f,
                   int* __restrict__ idx_r, float* __restrict__ w_r,
                   int* __restrict__ counts_f, int* __restrict__ counts_r)
{
  __shared__ float xs[Dm];
  __shared__ float proj[DSd];
  __shared__ float p[Nn];
  __shared__ float rv[Nn];
  __shared__ int   ri[Nn];
  __shared__ float tvs[TOPK];
  __shared__ int   tis[TOPK];

  const int t   = blockIdx.x;
  const int tid = threadIdx.x;
  const float* xr = x + (long long)t * Dm;
  for (int i = tid; i < Dm; i += 128) xs[i] = xr[i];
  __syncthreads();

  for (int route = 0; route < 2; ++route) {
    const float* W  = route ? W_rk : W_fk;
    const float* bb = route ? b_rk : b_fk;
    const float* em = route ? emb_rk : emb_fk;
    int*   idxb = route ? idx_r : idx_f;
    float* wb   = route ? w_r   : w_f;
    int*   cnts = route ? counts_r : counts_f;

    if (tid < DSd) {                    // proj = x @ W + b
      float s = bb[tid];
      for (int d = 0; d < Dm; ++d) s += xs[d] * W[d * DSd + tid];
      proj[tid] = s;
    }
    __syncthreads();

    {                                   // logits vs row-L2-normalized embeddings
      const float* e = em + tid * DSd;
      float ss = 0.f, dt = 0.f;
      for (int i = 0; i < DSd; ++i) { float v = e[i]; ss += v * v; dt += proj[i] * v; }
      p[tid] = dt / (sqrtf(ss) + 1e-8f);
    }
    __syncthreads();

    // softmax over 128
    rv[tid] = p[tid]; __syncthreads();
    for (int s = 64; s > 0; s >>= 1) { if (tid < s) rv[tid] = fmaxf(rv[tid], rv[tid + s]); __syncthreads(); }
    float mx = rv[0]; __syncthreads();
    float ev = __expf(p[tid] - mx);
    rv[tid] = ev; __syncthreads();
    for (int s = 64; s > 0; s >>= 1) { if (tid < s) rv[tid] += rv[tid + s]; __syncthreads(); }
    float sm = rv[0]; __syncthreads();
    p[tid] = ev / sm;
    __syncthreads();

    // top-4 via repeated argmax
    for (int it = 0; it < TOPK; ++it) {
      rv[tid] = p[tid]; ri[tid] = tid; __syncthreads();
      for (int s = 64; s > 0; s >>= 1) {
        if (tid < s && rv[tid + s] > rv[tid]) { rv[tid] = rv[tid + s]; ri[tid] = ri[tid + s]; }
        __syncthreads();
      }
      if (tid == 0) { tvs[it] = rv[0]; tis[it] = ri[0]; }
      __syncthreads();
      if (tid == ri[0]) p[tid] = -1.0f;
      __syncthreads();
    }
    if (tid == 0) {
      float s4 = tvs[0] + tvs[1] + tvs[2] + tvs[3] + 1e-8f;
      for (int j = 0; j < TOPK; ++j) {
        idxb[t * TOPK + j] = tis[j];
        wb[t * TOPK + j]   = tvs[j] / s4;
        atomicAdd(&cnts[tis[j]], 1);    // integer: order-independent
      }
    }
    __syncthreads();
  }
}

// ---------------------------------------------------------------- scan + deterministic bucket
__global__ void scan_kernel(const int* counts_f, const int* counts_r,
                            int* offs_f, int* offs_r)
{
  if (threadIdx.x == 0) {
    int a = 0;
    for (int i = 0; i < Nn; ++i) { offs_f[i] = a; a += counts_f[i]; }
    offs_f[Nn] = a;
    a = 0;
    for (int i = 0; i < Nn; ++i) { offs_r[i] = a; a += counts_r[i]; }
    offs_r[Nn] = a;
  }
}

// grid=(2), block=128: blockIdx selects route; thread n appends its matches in (t,j) order.
__global__ void bucket_kernel(const int* __restrict__ idx_f, const float* __restrict__ w_f,
                              const int* __restrict__ offs_f, int* __restrict__ btok_f, float* __restrict__ bw_f,
                              const int* __restrict__ idx_r, const float* __restrict__ w_r,
                              const int* __restrict__ offs_r, int* __restrict__ btok_r, float* __restrict__ bw_r)
{
  const int* idx  = blockIdx.x ? idx_r  : idx_f;
  const float* wv = blockIdx.x ? w_r    : w_f;
  const int* offs = blockIdx.x ? offs_r : offs_f;
  int* btok  = blockIdx.x ? btok_r : btok_f;
  float* bw  = blockIdx.x ? bw_r   : bw_f;
  int n = threadIdx.x;
  if (n >= Nn) return;
  int pos = offs[n];
  for (int i = 0; i < TOK * TOPK; ++i) {        // i encodes (token<<2)|slot
    if (idx[i] == n) { btok[pos] = i; bw[pos] = wv[i]; ++pos; }
  }
}

// ---------------------------------------------------------------- feature GEMM (grouped MoE)
// per expert n, per 16-token tile:  h_part[entry, 0:128] = w * (X[16,768] @ F_n[768,128])
__global__ __launch_bounds__(128)
void feature_gemm(const float* __restrict__ x, const float* __restrict__ f_know,
                  const int* __restrict__ btok, const float* __restrict__ bw,
                  const int* __restrict__ offs, const int* __restrict__ counts,
                  float* __restrict__ h_part)
{
  __shared__ __align__(32) __bf16 aF[2][32][16];        // [hi/lo][lane][elem]
  __shared__ __align__(32) __bf16 bF[2][8][32][16];     // [hi/lo][ntile][lane][elem]
  __shared__ int   toks[16];
  __shared__ float wts[16];
  __shared__ int   vld[16];

  const int n    = blockIdx.x;
  const int tid  = threadIdx.x;
  const int lane = tid & 31;
  const int wv   = tid >> 5;
  const float* Bn = f_know + (long long)n * Dm * Rr;
  const int cnt  = counts[n];
  const int base = offs[n];
  const int ntiles = (cnt + 15) >> 4;

  for (int tile = blockIdx.y; tile < ntiles; tile += gridDim.y) {
    __syncthreads();
    if (tid < 16) {
      int i = tile * 16 + tid;
      if (i < cnt) { toks[tid] = btok[base + i]; wts[tid] = bw[base + i]; vld[tid] = 1; }
      else         { toks[tid] = 0;              wts[tid] = 0.f;          vld[tid] = 0; }
    }
    __syncthreads();

    v8f acc0 = {}, acc1 = {};
    for (int k0 = 0; k0 < Dm; k0 += 32) {
      // A chunk 16x32 (split bf16 hi/lo), gathered token rows of x
      for (int idx = tid; idx < 512; idx += 128) {
        int m = idx >> 5, K = idx & 31;
        float v = vld[m] ? x[(long long)(toks[m] >> 2) * Dm + k0 + K] : 0.f;
        __bf16 hi = (__bf16)v;
        __bf16 lo = (__bf16)(v - (float)hi);
        int l = frag_lane(m, K), e = frag_elem(K);
        aF[0][l][e] = hi; aF[1][l][e] = lo;
      }
      // B chunk 32x128 of F_n
      for (int idx = tid; idx < 4096; idx += 128) {
        int K = idx >> 7, c = idx & 127;
        float v = Bn[(long long)(k0 + K) * Rr + c];
        __bf16 hi = (__bf16)v;
        __bf16 lo = (__bf16)(v - (float)hi);
        int tI = c >> 4, l = frag_lane(c & 15, K), e = frag_elem(K);
        bF[0][tI][l][e] = hi; bF[1][tI][l][e] = lo;
      }
      __syncthreads();
      v16bf aH = *(const v16bf*)&aF[0][lane][0];
      v16bf aL = *(const v16bf*)&aF[1][lane][0];
      {
        v16bf bH = *(const v16bf*)&bF[0][wv * 2 + 0][lane][0];
        v16bf bL = *(const v16bf*)&bF[1][wv * 2 + 0][lane][0];
        acc0 = wmma_bf16(aL, bH, acc0);
        acc0 = wmma_bf16(aH, bL, acc0);
        acc0 = wmma_bf16(aH, bH, acc0);
      }
      {
        v16bf bH = *(const v16bf*)&bF[0][wv * 2 + 1][lane][0];
        v16bf bL = *(const v16bf*)&bF[1][wv * 2 + 1][lane][0];
        acc1 = wmma_bf16(aL, bH, acc1);
        acc1 = wmma_bf16(aH, bL, acc1);
        acc1 = wmma_bf16(aH, bH, acc1);
      }
      __syncthreads();
    }
    // D tile layout: VGPR v -> M = v + 8*(lane/16), N = lane%16 (+16*ntile)
    int mbase = (lane >> 4) << 3;
    int n0 = (wv * 2 + 0) * 16 + (lane & 15);
    int n1 = (wv * 2 + 1) * 16 + (lane & 15);
    for (int v = 0; v < 8; ++v) {
      int m = mbase + v;
      if (vld[m]) {
        h_part[(long long)toks[m] * Rr + n0] = wts[m] * acc0[v];
        h_part[(long long)toks[m] * Rr + n1] = wts[m] * acc1[v];
      }
    }
  }
}

// ---------------------------------------------------------------- restore GEMM (grouped MoE)
// per expert n, 16-token tile, 256-col N-chunk: out_part[entry, nc0:nc0+256] = w*(H[16,128]@R_n[128,768])
__global__ __launch_bounds__(128)
void restore_gemm(const float* __restrict__ h, const float* __restrict__ r_know,
                  const int* __restrict__ btok, const float* __restrict__ bw,
                  const int* __restrict__ offs, const int* __restrict__ counts,
                  float* __restrict__ out_part)
{
  __shared__ __align__(32) __bf16 aF[2][32][16];
  __shared__ __align__(32) __bf16 bF[2][16][32][16];   // 16 N-subtiles (256 cols)
  __shared__ int   toks[16];
  __shared__ float wts[16];
  __shared__ int   vld[16];

  const int n    = blockIdx.x;
  const int tid  = threadIdx.x;
  const int lane = tid & 31;
  const int wv   = tid >> 5;
  const int nc0  = blockIdx.z * 256;
  const float* Bn = r_know + (long long)n * Rr * Dm;
  const int cnt  = counts[n];
  const int base = offs[n];
  const int ntiles = (cnt + 15) >> 4;

  for (int tile = blockIdx.y; tile < ntiles; tile += gridDim.y) {
    __syncthreads();
    if (tid < 16) {
      int i = tile * 16 + tid;
      if (i < cnt) { toks[tid] = btok[base + i]; wts[tid] = bw[base + i]; vld[tid] = 1; }
      else         { toks[tid] = 0;              wts[tid] = 0.f;          vld[tid] = 0; }
    }
    __syncthreads();

    v8f acc[4] = {};
    for (int k0 = 0; k0 < Rr; k0 += 32) {
      for (int idx = tid; idx < 512; idx += 128) {
        int m = idx >> 5, K = idx & 31;
        float v = vld[m] ? h[(long long)(toks[m] >> 2) * Rr + k0 + K] : 0.f;
        __bf16 hi = (__bf16)v;
        __bf16 lo = (__bf16)(v - (float)hi);
        int l = frag_lane(m, K), e = frag_elem(K);
        aF[0][l][e] = hi; aF[1][l][e] = lo;
      }
      for (int idx = tid; idx < 8192; idx += 128) {
        int K = idx >> 8, c = idx & 255;
        float v = Bn[(long long)(k0 + K) * Dm + nc0 + c];
        __bf16 hi = (__bf16)v;
        __bf16 lo = (__bf16)(v - (float)hi);
        int tI = c >> 4, l = frag_lane(c & 15, K), e = frag_elem(K);
        bF[0][tI][l][e] = hi; bF[1][tI][l][e] = lo;
      }
      __syncthreads();
      v16bf aH = *(const v16bf*)&aF[0][lane][0];
      v16bf aL = *(const v16bf*)&aF[1][lane][0];
#pragma unroll
      for (int q = 0; q < 4; ++q) {
        v16bf bH = *(const v16bf*)&bF[0][wv * 4 + q][lane][0];
        v16bf bL = *(const v16bf*)&bF[1][wv * 4 + q][lane][0];
        acc[q] = wmma_bf16(aL, bH, acc[q]);
        acc[q] = wmma_bf16(aH, bL, acc[q]);
        acc[q] = wmma_bf16(aH, bH, acc[q]);
      }
      __syncthreads();
    }
    int mbase = (lane >> 4) << 3;
#pragma unroll
    for (int q = 0; q < 4; ++q) {
      int d = nc0 + (wv * 4 + q) * 16 + (lane & 15);
      for (int v = 0; v < 8; ++v) {
        int m = mbase + v;
        if (vld[m]) out_part[(long long)toks[m] * Dm + d] = wts[m] * acc[q][v];
      }
    }
  }
}

// ---------------------------------------------------------------- host launcher
extern "C" void kernel_launch(void* const* d_in, const int* in_sizes, int n_in,
                              void* d_out, int out_size, void* d_ws, size_t ws_size,
                              hipStream_t stream)
{
  (void)in_sizes; (void)n_in; (void)out_size; (void)ws_size;
  const float* x    = (const float*)d_in[0];
  const float* f_kn = (const float*)d_in[1];
  const float* r_kn = (const float*)d_in[2];
  const float* W_fk = (const float*)d_in[3];
  const float* b_fk = (const float*)d_in[4];
  const float* W_rk = (const float*)d_in[5];
  const float* b_rk = (const float*)d_in[6];
  const float* e_fk = (const float*)d_in[7];
  const float* e_rk = (const float*)d_in[8];
  float* out = (float*)d_out;

  char* ws = (char*)d_ws;
  size_t off = 0;
  auto carve = [&](size_t bytes) -> void* {
    void* p = ws + off;
    off = (off + bytes + 255) & ~(size_t)255;
    return p;
  };
  int*   idx_f  = (int*)  carve((size_t)TOK * TOPK * 4);
  float* wgt_f  = (float*)carve((size_t)TOK * TOPK * 4);
  int*   idx_r  = (int*)  carve((size_t)TOK * TOPK * 4);
  float* wgt_r  = (float*)carve((size_t)TOK * TOPK * 4);
  int*   cnt_f  = (int*)  carve(Nn * 4);
  int*   cnt_r  = (int*)  carve(Nn * 4);
  int*   offs_f = (int*)  carve((Nn + 1) * 4);
  int*   offs_r = (int*)  carve((Nn + 1) * 4);
  int*   btok_f = (int*)  carve((size_t)TOK * TOPK * 4);
  float* bw_f   = (float*)carve((size_t)TOK * TOPK * 4);
  int*   btok_r = (int*)  carve((size_t)TOK * TOPK * 4);
  float* bw_r   = (float*)carve((size_t)TOK * TOPK * 4);
  float* h_part = (float*)carve((size_t)TOK * TOPK * Rr * 4);   // 4 MB
  float* hbuf   = (float*)carve((size_t)TOK * Rr * 4);          // 1 MB
  float* o_part = (float*)carve((size_t)TOK * TOPK * Dm * 4);   // 25 MB

  zero_u32<<<4, 64, 0, stream>>>((unsigned int*)cnt_f, Nn);
  zero_u32<<<4, 64, 0, stream>>>((unsigned int*)cnt_r, Nn);

  router_kernel<<<TOK, 128, 0, stream>>>(x, W_fk, b_fk, W_rk, b_rk, e_fk, e_rk,
                                         idx_f, wgt_f, idx_r, wgt_r, cnt_f, cnt_r);
  scan_kernel<<<1, 32, 0, stream>>>(cnt_f, cnt_r, offs_f, offs_r);
  bucket_kernel<<<2, 128, 0, stream>>>(idx_f, wgt_f, offs_f, btok_f, bw_f,
                                       idx_r, wgt_r, offs_r, btok_r, bw_r);

  feature_gemm<<<dim3(Nn, 32), 128, 0, stream>>>(x, f_kn, btok_f, bw_f, offs_f, cnt_f, h_part);
  reduce4_kernel<<<(TOK * Rr + 255) / 256, 256, 0, stream>>>(h_part, hbuf, Rr);

  restore_gemm<<<dim3(Nn, 32, 3), 128, 0, stream>>>(hbuf, r_kn, btok_r, bw_r, offs_r, cnt_r, o_part);
  reduce4_kernel<<<(TOK * Dm + 255) / 256, 256, 0, stream>>>(o_part, out, Dm);
}
